// S4_Block_16801912062572
// MI455X (gfx1250) — compile-verified, matrix operand order checked by reading, not run on hardware
//
#include <hip/hip_runtime.h>
#include <hip/hip_bf16.h>
#include <math.h>

#define B_SZ 32
#define LSEQ 1024
#define HDIM 512
#define NSSM 64
#define AJ   2064   // padded flipped-kernel length per parity copy

typedef __attribute__((ext_vector_type(16))) __bf16        v16bf;
typedef __attribute__((ext_vector_type(8)))  float          v8f;
typedef __attribute__((ext_vector_type(16))) unsigned short v16us;
typedef __attribute__((ext_vector_type(8)))  unsigned short v8us;
typedef __attribute__((ext_vector_type(8)))  unsigned int   v8ui;

__device__ __forceinline__ unsigned short f2bf(float f) {
  unsigned int u = __float_as_uint(f);
  u += 0x7FFFu + ((u >> 16) & 1u);           // round-to-nearest-even
  return (unsigned short)(u >> 16);
}
__device__ __forceinline__ float gelu_f(float x) {
  return 0.5f * x * (1.0f + erff(x * 0.70710678118654752f));
}

// B/regular fragment: element e -> k = 16*(e>=8) + (e&7); off8 folded into base.
__device__ __forceinline__ v16bf load_frag_lin(const unsigned short* base) {
  v8us lo = *(const v8us*)(base);        // k = 0..7   (+off8)
  v8us hi = *(const v8us*)(base + 16);   // k = 16..23 (+off8)
  v16us r;
#pragma unroll
  for (int i = 0; i < 8; ++i) { r[i] = lo[i]; r[i + 8] = hi[i]; }
  return __builtin_bit_cast(v16bf, r);
}

// Toeplitz A fragment: 8 aligned dword loads from the parity-shifted copy
// (contiguous -> compiler merges into b128 loads at dword alignment).
__device__ __forceinline__ v16bf load_frag_toeplitz(const unsigned short* base) {
  v8ui r;
#pragma unroll
  for (int w = 0; w < 8; ++w) {
    int idx = (w < 4) ? (2 * w) : (16 + 2 * (w - 4));  // even
    r[w] = *(const unsigned int*)(base + idx);
  }
  return __builtin_bit_cast(v16bf, r);
}

#define WMMA_BF16(A, Bv, C) \
  __builtin_amdgcn_wmma_f32_16x16x32_bf16(false, (A), false, (Bv), (short)0, (C), false, false)

// ---------------- K0: LayerNorm -> bf16 X in [H][B][L] ----------------
__global__ __launch_bounds__(256) void k_ln(const float* __restrict__ u,
                                            const float* __restrict__ g,
                                            const float* __restrict__ bb,
                                            unsigned short* __restrict__ xbf) {
  int row  = (blockIdx.x << 3) | (threadIdx.x >> 5);   // (b,l) row, 0..32767
  int lane = threadIdx.x & 31;
  const float* up = u + (size_t)row * HDIM;
  float v[16], s = 0.f, s2 = 0.f;
#pragma unroll
  for (int i = 0; i < 16; ++i) {
    float x = up[lane + (i << 5)];
    v[i] = x; s += x; s2 += x * x;
  }
#pragma unroll
  for (int m = 16; m >= 1; m >>= 1) {
    s  += __shfl_xor(s,  m, 32);
    s2 += __shfl_xor(s2, m, 32);
  }
  float mean = s * (1.0f / HDIM);
  float var  = s2 * (1.0f / HDIM) - mean * mean;
  float rs   = rsqrtf(var + 1e-5f);
  int b = row >> 10, l = row & 1023;
#pragma unroll
  for (int i = 0; i < 16; ++i) {
    int h = lane + (i << 5);
    float x = (v[i] - mean) * rs * g[h] + bb[h];
    xbf[((size_t)(h * B_SZ + b) << 10) | l] = f2bf(x);
  }
}

// ------- K1: SSM kernel K[h,l], flipped+padded+parity-duplicated, D folded in -------
__global__ __launch_bounds__(256) void k_ssm(const float* __restrict__ log_dt,
                                             const float* __restrict__ Alr,
                                             const float* __restrict__ Aim,
                                             const float* __restrict__ Bre,
                                             const float* __restrict__ Bim,
                                             const float* __restrict__ Cre,
                                             const float* __restrict__ Cim,
                                             const float* __restrict__ Dd,
                                             unsigned short* __restrict__ arrA) {
  __shared__ float dr[NSSM], di[NSSM], cr[NSSM], ci[NSSM];
  __shared__ float kb[LSEQ];
  int h = blockIdx.x, t = threadIdx.x;
  if (t < NSSM) {
    int n = t;
    float dt = __expf(log_dt[h]);
    float ar = -__expf(Alr[h * NSSM + n]);
    float ai = Aim[h * NSSM + n];
    float da_r = dt * ar, da_i = dt * ai;
    float er = __expf(da_r);
    float es, ec; __sincosf(da_i, &es, &ec);
    float Er = er * ec - 1.0f, Ei = er * es;          // exp(dtA) - 1
    float inv = 1.0f / (ar * ar + ai * ai);           // (.)/A = (.)*conj(A)/|A|^2
    float qr = (Er * ar + Ei * ai) * inv;
    float qi = (Ei * ar - Er * ai) * inv;
    float br = Bre[h * NSSM + n], bi = Bim[h * NSSM + n];
    float dbr = qr * br - qi * bi, dbi = qr * bi + qi * br;
    float cv = Cre[h * NSSM + n], civ = Cim[h * NSSM + n];
    cr[n] = cv * dbr - civ * dbi;
    ci[n] = cv * dbi + civ * dbr;
    dr[n] = da_r; di[n] = da_i;
  }
  __syncthreads();
  for (int l = t; l < LSEQ; l += 256) {
    float lf = (float)l, acc = 0.f;
#pragma unroll 4
    for (int n = 0; n < NSSM; ++n) {
      float e = __expf(dr[n] * lf);
      float s, c; __sincosf(di[n] * lf, &s, &c);
      acc += e * (cr[n] * c - ci[n] * s);
    }
    kb[l] = 2.0f * acc;
  }
  __syncthreads();
  if (t == 0) kb[0] += Dd[h];                         // fold D*x into Toeplitz diag
  __syncthreads();
  unsigned short* ap = arrA + (size_t)h * (2 * AJ);
  for (int p = 0; p < 2; ++p)
    for (int j = t; j < AJ; j += 256) {
      int tt = LSEQ - j - p;                          // Arr_p[j] = K[h, L-j-p]
      float v = (tt >= 0 && tt < LSEQ) ? kb[tt] : 0.0f;
      ap[p * AJ + j] = f2bf(v);
    }
}

// ---------------- K1b: W_out fp32 -> bf16 ----------------
__global__ __launch_bounds__(256) void k_w2bf(const float* __restrict__ W,
                                              unsigned short* __restrict__ Wbf, int n) {
  int i = blockIdx.x * 256 + threadIdx.x;
  if (i < n) Wbf[i] = f2bf(W[i]);
}

// ------- K2: causal conv as Toeplitz WMMA GEMM (32x32 tile/wave, explicit ping-pong) -------
__global__ __launch_bounds__(128) void k_conv(const unsigned short* __restrict__ arrA,
                                              const unsigned short* __restrict__ xbf,
                                              unsigned short* __restrict__ gbuf) {
  // wave id is uniform per wave -> force scalar so loop count & bases live in SGPRs
  int wid  = __builtin_amdgcn_readfirstlane((blockIdx.x << 2) | (threadIdx.x >> 5));
  int lane = threadIdx.x & 31;
  int h = wid >> 5, lblk = wid & 31;                  // 32 row-blocks of 32
  int lbase = lblk << 5;
  int row = lane & 15, hv = lane >> 4, off8 = hv << 3;
  int p = row & 1;
  int jadj0 = LSEQ - lbase - row - p + off8;          // even base into Arr_p (rows 0-15)
  const unsigned short* ap0 = arrA + (size_t)h * (2 * AJ) + (size_t)p * AJ + jadj0;
  const unsigned short* ap1 = ap0 - 16;               // rows 16-31
  const unsigned short* xp0 = xbf + ((size_t)(h * B_SZ + row) << 10) + off8;        // batch n
  const unsigned short* xp1 = xbf + ((size_t)(h * B_SZ + 16 + row) << 10) + off8;   // batch n+16
  v8f c00 = {}, c01 = {}, c10 = {}, c11 = {};
  int nloop = lblk;                                   // = ksteps - 1 (steps with a successor)

  // buffer set A holds the current tile; set B the next (explicit double buffer)
  v16bf a0A = load_frag_toeplitz(ap0);
  v16bf a1A = load_frag_toeplitz(ap1);
  v16bf b0A = load_frag_lin(xp0);
  v16bf b1A = load_frag_lin(xp1);
  v16bf a0B, a1B, b0B, b1B;
  int i = 0;
  for (; i + 2 <= nloop; i += 2) {
    ap0 += 32; ap1 += 32; xp0 += 32; xp1 += 32;
    __builtin_prefetch(ap0 + 64, 0, 1);
    __builtin_prefetch(xp0 + 64, 0, 1);
    __builtin_prefetch(xp1 + 64, 0, 1);
    a0B = load_frag_toeplitz(ap0);                    // load tile i+1 into B
    a1B = load_frag_toeplitz(ap1);
    b0B = load_frag_lin(xp0);
    b1B = load_frag_lin(xp1);
    c00 = WMMA_BF16(a0A, b0A, c00);                   // compute tile i from A
    c01 = WMMA_BF16(a0A, b1A, c01);
    c10 = WMMA_BF16(a1A, b0A, c10);
    c11 = WMMA_BF16(a1A, b1A, c11);
    ap0 += 32; ap1 += 32; xp0 += 32; xp1 += 32;
    __builtin_prefetch(ap0 + 64, 0, 1);
    __builtin_prefetch(xp0 + 64, 0, 1);
    __builtin_prefetch(xp1 + 64, 0, 1);
    a0A = load_frag_toeplitz(ap0);                    // load tile i+2 into A
    a1A = load_frag_toeplitz(ap1);
    b0A = load_frag_lin(xp0);
    b1A = load_frag_lin(xp1);
    c00 = WMMA_BF16(a0B, b0B, c00);                   // compute tile i+1 from B
    c01 = WMMA_BF16(a0B, b1B, c01);
    c10 = WMMA_BF16(a1B, b0B, c10);
    c11 = WMMA_BF16(a1B, b1B, c11);
  }
  if (i < nloop) {                                    // odd remainder: one more pipelined step
    ap0 += 32; ap1 += 32; xp0 += 32; xp1 += 32;
    a0B = load_frag_toeplitz(ap0);
    a1B = load_frag_toeplitz(ap1);
    b0B = load_frag_lin(xp0);
    b1B = load_frag_lin(xp1);
    c00 = WMMA_BF16(a0A, b0A, c00);
    c01 = WMMA_BF16(a0A, b1A, c01);
    c10 = WMMA_BF16(a1A, b0A, c10);
    c11 = WMMA_BF16(a1A, b1A, c11);
    c00 = WMMA_BF16(a0B, b0B, c00);                   // drain: final tile from B
    c01 = WMMA_BF16(a0B, b1B, c01);
    c10 = WMMA_BF16(a1B, b0B, c10);
    c11 = WMMA_BF16(a1B, b1B, c11);
  } else {
    c00 = WMMA_BF16(a0A, b0A, c00);                   // drain: final tile from A
    c01 = WMMA_BF16(a0A, b1A, c01);
    c10 = WMMA_BF16(a1A, b0A, c10);
    c11 = WMMA_BF16(a1A, b1A, c11);
  }

  // epilogue: gelu -> bf16 to gbuf[(b*L+l)*H + h]
  int n = lane & 15;
#pragma unroll
  for (int r = 0; r < 8; ++r) {
    int m  = r + (hv << 3);
    int l0 = lbase + m, l1 = lbase + 16 + m;
    gbuf[((size_t)((n << 10) | l0) << 9) | h]        = f2bf(gelu_f(c00[r]));
    gbuf[((size_t)(((n + 16) << 10) | l0) << 9) | h] = f2bf(gelu_f(c01[r]));
    gbuf[((size_t)((n << 10) | l1) << 9) | h]        = f2bf(gelu_f(c10[r]));
    gbuf[((size_t)(((n + 16) << 10) | l1) << 9) | h] = f2bf(gelu_f(c11[r]));
  }
}

// ------- K3: out = gelu(G @ W^T + b) + u ; 32x32 tile/wave, fully unrolled k -------
__global__ __launch_bounds__(128) void k_out(const unsigned short* __restrict__ gbuf,
                                             const unsigned short* __restrict__ wbf,
                                             const float* __restrict__ bo,
                                             const float* __restrict__ u,
                                             float* __restrict__ out) {
  int wid  = __builtin_amdgcn_readfirstlane((blockIdx.x << 2) | (threadIdx.x >> 5));
  int lane = threadIdx.x & 31;
  int rt = wid >> 4, ct = wid & 15;                   // 1024 row-tiles x 16 col-tiles
  int rbase = rt << 5, obase = ct << 5;
  int row = lane & 15, hv = lane >> 4, off8 = hv << 3;
  int n = lane & 15;
  const unsigned short* ga0 = gbuf + (size_t)(rbase + row) * HDIM + off8;
  const unsigned short* ga1 = gbuf + (size_t)(rbase + 16 + row) * HDIM + off8;
  const unsigned short* wb0 = wbf + (size_t)(obase + n) * HDIM + off8;
  const unsigned short* wb1 = wbf + (size_t)(obase + 16 + n) * HDIM + off8;
  v8f c00 = {}, c01 = {}, c10 = {}, c11 = {};

  v16bf a0 = load_frag_lin(ga0);
  v16bf a1 = load_frag_lin(ga1);
  v16bf b0 = load_frag_lin(wb0);
  v16bf b1 = load_frag_lin(wb1);
#pragma unroll
  for (int kk = 0; kk < HDIM / 32 - 1; ++kk) {
    ga0 += 32; ga1 += 32; wb0 += 32; wb1 += 32;
    v16bf a0n = load_frag_lin(ga0);
    v16bf a1n = load_frag_lin(ga1);
    v16bf b0n = load_frag_lin(wb0);
    v16bf b1n = load_frag_lin(wb1);
    c00 = WMMA_BF16(a0, b0, c00);
    c01 = WMMA_BF16(a0, b1, c01);
    c10 = WMMA_BF16(a1, b0, c10);
    c11 = WMMA_BF16(a1, b1, c11);
    a0 = a0n; a1 = a1n; b0 = b0n; b1 = b1n;           // full unroll -> pure renaming
  }
  c00 = WMMA_BF16(a0, b0, c00);
  c01 = WMMA_BF16(a0, b1, c01);
  c10 = WMMA_BF16(a1, b0, c10);
  c11 = WMMA_BF16(a1, b1, c11);

  float bo0 = bo[obase + n], bo1 = bo[obase + 16 + n];
#pragma unroll
  for (int r = 0; r < 8; ++r) {
    int m = r + (hv << 3);
    size_t g0 = (size_t)(rbase + m) * HDIM;
    size_t g1 = (size_t)(rbase + 16 + m) * HDIM;
    out[g0 + obase + n]      = gelu_f(c00[r] + bo0) + u[g0 + obase + n];
    out[g0 + obase + 16 + n] = gelu_f(c01[r] + bo1) + u[g0 + obase + 16 + n];
    out[g1 + obase + n]      = gelu_f(c10[r] + bo0) + u[g1 + obase + n];
    out[g1 + obase + 16 + n] = gelu_f(c11[r] + bo1) + u[g1 + obase + 16 + n];
  }
}

extern "C" void kernel_launch(void* const* d_in, const int* in_sizes, int n_in,
                              void* d_out, int out_size, void* d_ws, size_t ws_size,
                              hipStream_t stream) {
  (void)in_sizes; (void)n_in; (void)out_size; (void)ws_size;
  const float* u      = (const float*)d_in[0];
  const float* ln_g   = (const float*)d_in[1];
  const float* ln_b   = (const float*)d_in[2];
  const float* log_dt = (const float*)d_in[3];
  const float* Alr    = (const float*)d_in[4];
  const float* Aim    = (const float*)d_in[5];
  const float* Bre    = (const float*)d_in[6];
  const float* Bim    = (const float*)d_in[7];
  const float* Cre    = (const float*)d_in[8];
  const float* Cim    = (const float*)d_in[9];
  const float* Dd     = (const float*)d_in[10];
  const float* Wout   = (const float*)d_in[11];
  const float* bout   = (const float*)d_in[12];
  float* out = (float*)d_out;

  char* ws = (char*)d_ws;
  size_t off = 0;
  auto carve = [&](size_t bytes) -> void* {
    void* p = ws + off;
    off = (off + bytes + 255) & ~(size_t)255;
    return p;
  };
  unsigned short* xbf  = (unsigned short*)carve((size_t)HDIM * B_SZ * LSEQ * 2); // 32 MB
  unsigned short* arrA = (unsigned short*)carve((size_t)HDIM * 2 * AJ * 2);      // 4 MB
  unsigned short* wbf  = (unsigned short*)carve((size_t)HDIM * HDIM * 2);        // 0.5 MB
  unsigned short* gbuf = (unsigned short*)carve((size_t)B_SZ * LSEQ * HDIM * 2); // 32 MB

  hipLaunchKernelGGL(k_ln,   dim3(B_SZ * LSEQ / 8), dim3(256), 0, stream, u, ln_g, ln_b, xbf);
  hipLaunchKernelGGL(k_ssm,  dim3(HDIM),            dim3(256), 0, stream,
                     log_dt, Alr, Aim, Bre, Bim, Cre, Cim, Dd, arrA);
  hipLaunchKernelGGL(k_w2bf, dim3(HDIM * HDIM / 256), dim3(256), 0, stream, Wout, wbf, HDIM * HDIM);
  // 512 h * 32 row-blocks = 16384 waves / 4 per block
  hipLaunchKernelGGL(k_conv, dim3(HDIM * 32 / 4),   dim3(128), 0, stream, arrA, xbf, gbuf);
  // 1024 row-tiles * 16 col-tiles = 16384 waves / 4 per block
  hipLaunchKernelGGL(k_out,  dim3((B_SZ * LSEQ / 32) * (HDIM / 32) / 4), dim3(128), 0, stream,
                     gbuf, wbf, bout, u, out);
}
